// EfficientWindowAttention_7352984011439
// MI455X (gfx1250) — matrix operand order repaired and checked by reading, not compiled, not run on hardware
//
#include <hip/hip_runtime.h>
#include <hip/hip_bf16.h>

// ---------------------------------------------------------------------------
// Fused 7x7 window attention for MI455X (gfx1250, wave32, WMMA bf16).
// One workgroup per window; QKV GEMM -> attention -> proj all in LDS.
// ---------------------------------------------------------------------------

#define WSZ   7
#define NTOK  49
#define NPAD  64
#define CDIM  128
#define NHEAD 4
#define HDIM  32
#define IMGH  448
#define IMGW  448
#define WIN_PER_IMG 4096     // (448/7)^2
#define NWIN  16384          // 4 * 4096

typedef __bf16 bf16_t;
typedef __attribute__((ext_vector_type(16))) __bf16 bf16x16;
typedef __attribute__((ext_vector_type(8)))  __bf16 bf16x8;
typedef __attribute__((ext_vector_type(8)))  float  f32x8;

union ABf { bf16x16 v; bf16x8 h[2]; };

__device__ __forceinline__ bf16_t f2bf(float f) {
  union { float f; unsigned u; } c; c.f = f;
  unsigned u = c.u;
  u += 0x7fffu + ((u >> 16) & 1u);            // round-to-nearest-even
  union { unsigned short s; bf16_t b; } r; r.s = (unsigned short)(u >> 16);
  return r.b;
}

// -------- workspace layout (bytes) --------
#define WS_QKVW  0           // w_qkv transposed, bf16 [384][128]  = 98304 B
#define WS_PROJW 98304       // w_proj transposed, bf16 [128][128] = 32768 B
#define WS_BIAS  131072      // padded bias matrix f32 [64][64]    = 16384 B

// -------- LDS layout (bytes); strides in bf16 elements --------
#define XS 136               // x / out tile row stride (128 + 8 pad)
#define QS 40                // q/k row stride (32 + 8 pad)
#define VS 72                // vT row stride (64 + 8 pad)
#define AS 72                // attn row stride
#define OFF_X 0              // x tile [64][136] bf16 = 17408 (reused as outbuf)
#define OFF_Q 17408          // q [4][64][40] bf16 = 20480
#define OFF_K 37888          // k [4][64][40] bf16 = 20480
#define OFF_V 58368          // vT [4][32][72] bf16 = 18432
#define OFF_A 76800          // attn [2 heads][64][72] bf16 = 18432
#define SMEM_BYTES 95232

// ---------------------------------------------------------------------------
// Prep: transpose weights to N-major bf16; build padded bias matrix.
// ---------------------------------------------------------------------------
__global__ void ewa_prep_kernel(const float* __restrict__ w_qkv,
                                const float* __restrict__ w_proj,
                                const float* __restrict__ bias_table,
                                unsigned char* __restrict__ ws) {
  bf16_t* wqkvT  = (bf16_t*)(ws + WS_QKVW);
  bf16_t* wprojT = (bf16_t*)(ws + WS_PROJW);
  float*  biasM  = (float*)(ws + WS_BIAS);
  int tid = blockIdx.x * blockDim.x + threadIdx.x;
  int stride = gridDim.x * blockDim.x;
  for (int i = tid; i < 384 * 128; i += stride) {
    int n = i >> 7, k = i & 127;
    wqkvT[n * 128 + k] = f2bf(w_qkv[k * 384 + n]);
  }
  for (int i = tid; i < 128 * 128; i += stride) {
    int n = i >> 7, k = i & 127;
    wprojT[n * 128 + k] = f2bf(w_proj[k * 128 + n]);
  }
  for (int i = tid; i < NPAD * NPAD; i += stride) {
    int m = i >> 6, n = i & 63;
    float v = -1e30f;                          // mask padded cols/rows
    if (m < NTOK && n < NTOK) {
      int rm = m / WSZ, cm = m % WSZ, rn = n / WSZ, cn = n % WSZ;
      v = bias_table[rm - rn + WSZ - 1] + bias_table[cm - cn + WSZ - 1];
    }
    biasM[i] = v;
  }
}

// ---------------------------------------------------------------------------
// Fused window attention kernel.
// ---------------------------------------------------------------------------
__global__ __launch_bounds__(256)
void EfficientWindowAttention_7352984011439_kernel(
    const float* __restrict__ x,
    const float* __restrict__ b_qkv,
    const float* __restrict__ b_proj,
    const unsigned char* __restrict__ ws,
    float* __restrict__ out) {
  __shared__ __align__(16) unsigned char smem[SMEM_BYTES];
  bf16_t* xA  = (bf16_t*)(smem + OFF_X);   // A-tile of x; later: pre-proj out
  bf16_t* ob  = (bf16_t*)(smem + OFF_X);
  bf16_t* qs  = (bf16_t*)(smem + OFF_Q);
  bf16_t* ksh = (bf16_t*)(smem + OFF_K);
  bf16_t* vT  = (bf16_t*)(smem + OFF_V);
  bf16_t* at  = (bf16_t*)(smem + OFF_A);

  const bf16_t* wqkvT  = (const bf16_t*)(ws + WS_QKVW);
  const bf16_t* wprojT = (const bf16_t*)(ws + WS_PROJW);
  const float*  biasM  = (const float*)(ws + WS_BIAS);

  const int widx = blockIdx.x;
  const int bimg = widx >> 12;
  const int rem  = widx & 4095;
  const int h0   = (rem >> 6) * WSZ;
  const int w0   = (rem & 63) * WSZ;

  const int tid  = threadIdx.x;
  const int lane = tid & 31;
  const int wv   = tid >> 5;      // wave id 0..7
  const int lmod = lane & 15;
  const int lhi  = lane >> 4;     // 0 or 1

  // ---------- phase 0: load window -> bf16 LDS (zero-pad rows 49..63) ------
  for (int i = tid; i < NPAD * CDIM; i += 256) {
    int t = i >> 7, c = i & 127;
    float v = 0.0f;
    if (t < NTOK) {
      int hh = h0 + t / WSZ, ww = w0 + t % WSZ;
      v = x[(((size_t)bimg * IMGH + hh) * IMGW + ww) * CDIM + c];
    }
    xA[t * XS + c] = f2bf(v);
  }
  __syncthreads();

  // ---------- phase 1: QKV = xw @ w_qkv + b_qkv (96 tiles over 8 waves) ----
  for (int tile = wv; tile < 96; tile += 8) {
    const int mt = tile & 3, nt = tile >> 2;
    const int arow = mt * 16 + lmod;
    const int ncol = nt * 16 + lmod;
    f32x8 acc = {};
#pragma unroll
    for (int kc = 0; kc < 4; ++kc) {
      ABf a, b;
      const int kA = kc * 32 + lhi * 8;
      a.h[0] = *(const bf16x8*)(xA + arow * XS + kA);
      a.h[1] = *(const bf16x8*)(xA + arow * XS + kA + 16);
      const int kB = kc * 32 + lhi * 8;
      const bf16_t* bp = wqkvT + ncol * CDIM + kB;
      b.h[0] = *(const bf16x8*)(bp);
      b.h[1] = *(const bf16x8*)(bp + 16);
      acc = __builtin_amdgcn_wmma_f32_16x16x32_bf16(
          false, a.v, false, b.v, (short)0, acc, false, false);
    }
    const float bq = b_qkv[ncol];
    const int tokbase = mt * 16 + lhi * 8;
    if (ncol < 128) {                       // Q: [head][token][hd]
      const int head = ncol >> 5, d = ncol & 31;
      bf16_t* dst = qs + head * NPAD * QS + d;
#pragma unroll
      for (int r = 0; r < 8; ++r) dst[(tokbase + r) * QS] = f2bf(acc[r] + bq);
    } else if (ncol < 256) {                // K: [head][token][hd]
      const int c2 = ncol - 128, head = c2 >> 5, d = c2 & 31;
      bf16_t* dst = ksh + head * NPAD * QS + d;
#pragma unroll
      for (int r = 0; r < 8; ++r) dst[(tokbase + r) * QS] = f2bf(acc[r] + bq);
    } else {                                // V transposed: [head][hd][token]
      const int c2 = ncol - 256, head = c2 >> 5, d = c2 & 31;
      bf16_t* dst = vT + (head * HDIM + d) * VS + tokbase;
#pragma unroll
      for (int r = 0; r < 8; ++r) dst[r] = f2bf(acc[r] + bq);
    }
  }
  __syncthreads();

  // ---------- phase 2: attention, two heads at a time ----------------------
  const float scale = 0.17677669529663687f;  // 1/sqrt(32)
  const int hl = wv >> 2;                    // head slot within pair
  const int mt = wv & 3;                     // this wave's row tile
  const int arow = mt * 16 + lmod;
  const int rowbase = mt * 16 + lhi * 8;

  for (int hp = 0; hp < 2; ++hp) {
    const int head = hp * 2 + hl;
    const bf16_t* qb = qs + head * NPAD * QS;
    const bf16_t* kb = ksh + head * NPAD * QS;

    // logits = q @ k^T for this wave's 16 rows, all 64 cols
    f32x8 lg[4];
#pragma unroll
    for (int nt = 0; nt < 4; ++nt) {
      f32x8 acc = {};
      ABf a, b;
      const int kA = lhi * 8;
      a.h[0] = *(const bf16x8*)(qb + arow * QS + kA);
      a.h[1] = *(const bf16x8*)(qb + arow * QS + kA + 16);
      const int ntok = nt * 16 + lmod;
      const int kB = lhi * 8;
      b.h[0] = *(const bf16x8*)(kb + ntok * QS + kB);
      b.h[1] = *(const bf16x8*)(kb + ntok * QS + kB + 16);
      lg[nt] = __builtin_amdgcn_wmma_f32_16x16x32_bf16(
          false, a.v, false, b.v, (short)0, acc, false, false);
    }

    // scale + bias + row softmax (rows live in 16-lane halves)
#pragma unroll
    for (int r = 0; r < 8; ++r) {
      const int m = rowbase + r;
      float v0 = lg[0][r] * scale + biasM[m * 64 + 0 * 16 + lmod];
      float v1 = lg[1][r] * scale + biasM[m * 64 + 1 * 16 + lmod];
      float v2 = lg[2][r] * scale + biasM[m * 64 + 2 * 16 + lmod];
      float v3 = lg[3][r] * scale + biasM[m * 64 + 3 * 16 + lmod];
      float mx = fmaxf(fmaxf(v0, v1), fmaxf(v2, v3));
#pragma unroll
      for (int off = 1; off < 16; off <<= 1) mx = fmaxf(mx, __shfl_xor(mx, off, 32));
      float e0 = __expf(v0 - mx), e1 = __expf(v1 - mx);
      float e2 = __expf(v2 - mx), e3 = __expf(v3 - mx);
      float s = e0 + e1 + e2 + e3;
#pragma unroll
      for (int off = 1; off < 16; off <<= 1) s += __shfl_xor(s, off, 32);
      const float inv = 1.0f / s;
      bf16_t* ar = at + (hl * NPAD + m) * AS;
      ar[0 * 16 + lmod] = f2bf(e0 * inv);
      ar[1 * 16 + lmod] = f2bf(e1 * inv);
      ar[2 * 16 + lmod] = f2bf(e2 * inv);
      ar[3 * 16 + lmod] = f2bf(e3 * inv);
    }
    // attn rows are wave-local -> no workgroup barrier needed.

    // out_head = attn @ v  (2 x 16-col tiles of head_dim 32)
#pragma unroll
    for (int ndt = 0; ndt < 2; ++ndt) {
      f32x8 acc = {};
      const int dcol = ndt * 16 + lmod;
#pragma unroll
      for (int kc = 0; kc < 2; ++kc) {
        ABf a, b;
        const int kA = kc * 32 + lhi * 8;
        const bf16_t* ap = at + (hl * NPAD + arow) * AS + kA;
        a.h[0] = *(const bf16x8*)(ap);
        a.h[1] = *(const bf16x8*)(ap + 16);
        const int kB = kc * 32 + lhi * 8;
        const bf16_t* vp = vT + (head * HDIM + dcol) * VS + kB;
        b.h[0] = *(const bf16x8*)(vp);
        b.h[1] = *(const bf16x8*)(vp + 16);
        acc = __builtin_amdgcn_wmma_f32_16x16x32_bf16(
            false, a.v, false, b.v, (short)0, acc, false, false);
      }
      const int ocol = head * HDIM + dcol;
#pragma unroll
      for (int r = 0; r < 8; ++r)
        ob[(rowbase + r) * XS + ocol] = f2bf(acc[r]);
    }
  }
  __syncthreads();

  // ---------- phase 3: proj = ob @ w_proj + b_proj, scatter to output ------
  for (int tile = wv; tile < 32; tile += 8) {
    const int mt3 = tile & 3, nt = tile >> 2;
    const int arow3 = mt3 * 16 + lmod;
    const int ncol = nt * 16 + lmod;
    f32x8 acc = {};
#pragma unroll
    for (int kc = 0; kc < 4; ++kc) {
      ABf a, b;
      const int kA = kc * 32 + lhi * 8;
      const bf16_t* ap = ob + arow3 * XS + kA;
      a.h[0] = *(const bf16x8*)(ap);
      a.h[1] = *(const bf16x8*)(ap + 16);
      const int kB = kc * 32 + lhi * 8;
      const bf16_t* bp = wprojT + ncol * CDIM + kB;
      b.h[0] = *(const bf16x8*)(bp);
      b.h[1] = *(const bf16x8*)(bp + 16);
      acc = __builtin_amdgcn_wmma_f32_16x16x32_bf16(
          false, a.v, false, b.v, (short)0, acc, false, false);
    }
    const float bp2 = b_proj[ncol];
    const int tokbase = mt3 * 16 + lhi * 8;
#pragma unroll
    for (int r = 0; r < 8; ++r) {
      const int t = tokbase + r;
      if (t < NTOK) {
        const int hh = h0 + t / WSZ, ww = w0 + t % WSZ;
        out[(((size_t)bimg * IMGH + hh) * IMGW + ww) * CDIM + ncol] = acc[r] + bp2;
      }
    }
  }
}

// ---------------------------------------------------------------------------
extern "C" void kernel_launch(void* const* d_in, const int* in_sizes, int n_in,
                              void* d_out, int out_size, void* d_ws, size_t ws_size,
                              hipStream_t stream) {
  const float* x          = (const float*)d_in[0];
  const float* w_qkv      = (const float*)d_in[1];
  const float* b_qkv      = (const float*)d_in[2];
  const float* w_proj     = (const float*)d_in[3];
  const float* b_proj     = (const float*)d_in[4];
  const float* bias_table = (const float*)d_in[5];
  float* out = (float*)d_out;
  unsigned char* ws = (unsigned char*)d_ws;

  ewa_prep_kernel<<<64, 256, 0, stream>>>(w_qkv, w_proj, bias_table, ws);
  EfficientWindowAttention_7352984011439_kernel<<<NWIN, 256, 0, stream>>>(
      x, b_qkv, b_proj, ws, out);
}
